// DGCNN_MS_70858370449935
// MI455X (gfx1250) — matrix operand order repaired and checked by reading, not compiled, not run on hardware
//
#include <hip/hip_runtime.h>
#include <hip/hip_bf16.h>
#include <math.h>

// ---------------------------------------------------------------------------
// DGCNN forward for MI455X (gfx1250, wave32, WMMA).
// All GEMMs (EdgeConv, Gram/kNN, W6, decoder) use v_wmma_f32_16x16x32_bf16
// with f32 accumulation; BN + LeakyReLU + max-over-k fused into epilogues.
// EdgeConv neighbor gather uses GLOBAL_LOAD_ASYNC_TO_LDS_B128 (ASYNCcnt).
// ---------------------------------------------------------------------------

typedef __bf16 bf16;
typedef __attribute__((ext_vector_type(16))) __bf16 v16bf;
typedef __attribute__((ext_vector_type(8)))  __bf16 v8bf;
typedef __attribute__((ext_vector_type(8)))  float  v8f;

#define BN_SCALE_C 0.9999950000374997f   // 1/sqrt(1+1e-5)
#define NEG_INF_F  (-__builtin_inff())

// ---- WMMA fragment helpers (layouts per CDNA5 ISA 7.12.2) ------------------
// A-matrix 16x32 bf16: lane L (row = L%16) holds K = q..q+7 and q+16..q+23,
// q = (L<16 ? 0 : 8).  Two 16B contiguous runs.
static __device__ __forceinline__ v16bf load_a_frag(const bf16* p, int lane) {
  int q = (lane < 16) ? 0 : 8;
  union { v16bf v; v8bf h[2]; } u;
  u.h[0] = *(const v8bf*)(p + q);
  u.h[1] = *(const v8bf*)(p + q + 16);
  return u.v;
}
// B-matrix 32x16 bf16: lane L (col = L%16) holds K = q..q+7 and q+16..q+23, q=(L<16?0:8).
static __device__ __forceinline__ v16bf load_b_frag(const bf16* p, int lane) {
  int q = (lane < 16) ? 0 : 8;
  union { v16bf v; v8bf h[2]; } u;
  u.h[0] = *(const v8bf*)(p + q);
  u.h[1] = *(const v8bf*)(p + q + 16);
  return u.v;
}
static __device__ __forceinline__ v8f wmma_bf16(v16bf a, v16bf b, v8f c) {
  return __builtin_amdgcn_wmma_f32_16x16x32_bf16(false, a, false, b,
                                                 (short)0, c, false, false);
}

// ---- async global->LDS helpers (CDNA5 ISA ch.10 / 08_async_tensor) ---------
static __device__ __forceinline__ void async_ld_b128(unsigned int lds_off,
                                                     unsigned long long gaddr) {
  asm volatile("global_load_async_to_lds_b128 %0, %1, off"
               :: "v"(lds_off), "v"(gaddr) : "memory");
}
static __device__ __forceinline__ void wait_async0() {
  asm volatile("s_wait_asynccnt 0x0" ::: "memory");
}

// ---- 1) f32 -> zero-padded bf16 (weights and per-layer feature staging) ----
__global__ void convert_pad_bf16(const float* __restrict__ src, int ldsrc,
                                 bf16* __restrict__ dst, int lddst,
                                 int rows, int cols) {
  long i = (long)blockIdx.x * blockDim.x + threadIdx.x;
  long total = (long)rows * lddst;
  if (i >= total) return;
  int r = (int)(i / lddst), c = (int)(i % lddst);
  float v = (c < cols) ? src[(long)r * ldsrc + c] : 0.f;
  dst[i] = (bf16)v;
}

// ---- 2) x (B,C,N) -> h0 (B,N,C) --------------------------------------------
__global__ void transpose_x_kernel(const float* __restrict__ x,
                                   float* __restrict__ h0,
                                   int B, int C, int N) {
  long i = (long)blockIdx.x * blockDim.x + threadIdx.x;
  long total = (long)B * C * N;
  if (i >= total) return;
  int n = (int)(i % N);
  int c = (int)((i / N) % C);
  int b = (int)(i / ((long)N * C));
  h0[((long)b * N + n) * C + c] = x[i];
}

// ---- 3) Gram matrix G[b] = H H^T via WMMA (one 16x16 tile per wave) --------
__global__ void gram_kernel(const bf16* __restrict__ H, int ldh,
                            float* __restrict__ G, int N, int Kpad) {
  int wave = threadIdx.x >> 5;
  int lane = threadIdx.x & 31;
  int b = blockIdx.y;
  int tile = blockIdx.x * 8 + wave;          // N/16=64 -> 64*64 tiles / batch
  int tn = tile >> 6, tm = tile & 63;
  const bf16* Hb = H + (long)b * N * ldh;
  int arow = tn * 16 + (lane & 15);
  int bcol = tm * 16 + (lane & 15);
  v8f acc = {0.f,0.f,0.f,0.f,0.f,0.f,0.f,0.f};
  for (int k = 0; k < Kpad; k += 32) {
    v16bf a  = load_a_frag(Hb + (long)arow * ldh + k, lane);
    v16bf bb = load_b_frag(Hb + (long)bcol * ldh + k, lane);
    acc = wmma_bf16(a, bb, acc);
  }
  float* Gb = G + (long)b * N * N;
  int col   = tm * 16 + (lane & 15);
  int rbase = tn * 16 + ((lane < 16) ? 0 : 8);
#pragma unroll
  for (int i = 0; i < 8; i++)
    Gb[(long)(rbase + i) * N + col] = acc[i];
}

// ---- 4) top-K=20 of pd[m] = 2*G[n][m] - G[n][n] - G[m][m] ------------------
__global__ void topk_kernel(const float* __restrict__ G, int N,
                            int* __restrict__ idx, int K) {
  int b = blockIdx.x / N, n = blockIdx.x % N;
  const float* Gb  = G + (long)b * N * N;
  const float* row = Gb + (long)n * N;
  __shared__ float vals[1024];
  __shared__ float rmax[256];
  __shared__ int   rarg[256];
  float xn = row[n];
  int t = threadIdx.x;
  for (int m = t; m < N; m += blockDim.x)
    vals[m] = 2.f * row[m] - xn - Gb[(long)m * N + m];
  __syncthreads();
  for (int sel = 0; sel < K; ++sel) {
    float best = NEG_INF_F; int barg = 0;
    for (int m = t; m < N; m += blockDim.x) {
      float v = vals[m];
      if (v > best) { best = v; barg = m; }
    }
    rmax[t] = best; rarg[t] = barg;
    __syncthreads();
    for (int s = blockDim.x >> 1; s > 0; s >>= 1) {
      if (t < s && rmax[t + s] > rmax[t]) { rmax[t] = rmax[t + s]; rarg[t] = rarg[t + s]; }
      __syncthreads();
    }
    if (t == 0) { idx[(long)blockIdx.x * K + sel] = rarg[0]; vals[rarg[0]] = NEG_INF_F; }
    __syncthreads();
  }
}

// ---- 5) EdgeConv: async gather -> LDS -> bf16 -> WMMA -> BN/lrelu -> max ---
// LDS: [ rowsF : (K+1)*C f32 raw rows (20 nbrs + center) | smemB : 32 x Kpad bf16 ]
__global__ void edgeconv_kernel(const float* __restrict__ Hf, int ldh, int C,
                                const int* __restrict__ idx, int K,
                                const bf16* __restrict__ Wb, int ldw, int O,
                                const float* __restrict__ g,
                                const float* __restrict__ beta,
                                float* __restrict__ out, int ldo, int N,
                                int rfbytes, int use_async) {
  int bn = blockIdx.x;                 // b*N + n
  int b = bn / N, n = bn % N;
  int Kpad = ldw;                      // round32(2C)
  extern __shared__ char smembase[];
  float* rowsF = (float*)smembase;                 // (K+1) x C f32
  bf16*  smemB = (bf16*)(smembase + rfbytes);      // 32 x Kpad bf16
  const float* Hb  = Hf + (long)b * N * ldh;
  const int*   nb  = idx + (long)bn * K;
  int t = threadIdx.x;

  for (int i = t; i < 32 * Kpad; i += blockDim.x) smemB[i] = (bf16)0.f;

  if (use_async) {
    // async DMA of (K+1) rows of C floats, 16B granules, tracked by ASYNCcnt
    int nvec = (K + 1) * (C >> 2);
    for (int j = t; j < nvec; j += blockDim.x) {
      int row = j / (C >> 2), c4 = (j % (C >> 2)) << 2;
      int grow = (row < K) ? nb[row] : n;
      unsigned long long ga =
          (unsigned long long)(uintptr_t)(Hb + (long)grow * ldh + c4);
      unsigned int lo =
          (unsigned int)(uintptr_t)(rowsF + row * C + c4);  // low32 = LDS addr
      async_ld_b128(lo, ga);
    }
    wait_async0();
  } else {
    for (int j = t; j < (K + 1) * C; j += blockDim.x) {
      int row = j / C, c = j % C;
      int grow = (row < K) ? nb[row] : n;
      rowsF[row * C + c] = Hb[(long)grow * ldh + c];
    }
  }
  __syncthreads();

  int twoC = 2 * C;
  const float* ctr = rowsF + K * C;
  for (int e = t; e < K * twoC; e += blockDim.x) {
    int k = e / twoC, c = e % twoC;
    float cv = (c < C) ? (rowsF[k * C + c] - ctr[c]) : ctr[c - C];
    smemB[k * Kpad + c] = (bf16)cv;
  }
  __syncthreads();

  int wave = t >> 5, lane = t & 31;
  int ntiles = O >> 4;
  for (int nt = wave; nt < ntiles; nt += 8) {
    v8f acc0 = {0.f,0.f,0.f,0.f,0.f,0.f,0.f,0.f};
    v8f acc1 = {0.f,0.f,0.f,0.f,0.f,0.f,0.f,0.f};
    int col = nt * 16 + (lane & 15);
    const bf16* wcol  = Wb + (long)col * ldw;
    const bf16* arow0 = smemB + (lane & 15) * Kpad;
    const bf16* arow1 = smemB + (16 + (lane & 15)) * Kpad;
    for (int kk = 0; kk < Kpad; kk += 32) {
      v16bf bfr = load_b_frag(wcol + kk, lane);
      v16bf a0  = load_a_frag(arow0 + kk, lane);
      v16bf a1  = load_a_frag(arow1 + kk, lane);
      acc0 = wmma_bf16(a0, bfr, acc0);
      acc1 = wmma_bf16(a1, bfr, acc1);
    }
    float gs = g[col] * BN_SCALE_C, bb = beta[col];
    float m0 = NEG_INF_F, m1 = NEG_INF_F;
#pragma unroll
    for (int i = 0; i < 8; i++) {           // k rows 0..15 (all valid)
      float v = acc0[i] * gs + bb;
      v = (v >= 0.f) ? v : 0.2f * v;
      m0 = fmaxf(m0, v);
    }
    if (lane < 16) {                        // k rows 16..19 valid only
#pragma unroll
      for (int i = 0; i < 4; i++) {
        float v = acc1[i] * gs + bb;
        v = (v >= 0.f) ? v : 0.2f * v;
        m1 = fmaxf(m1, v);
      }
    }
    m0 = fmaxf(m0, __shfl_xor(m0, 16, 32));
    m1 = fmaxf(m1, __shfl_xor(m1, 16, 32));
    float mv = fmaxf(m0, m1);
    if (lane < 16) out[(long)bn * ldo + col] = mv;
  }
}

// ---- 6) Generic GEMM  Y = epi(X[M,K] * Wb[O,Kpad]^T) -----------------------
// Two 16-col tiles per wave: A-fragment (LDS) reused across two B streams.
// flags: 1=bias, 2=bn(g,beta), 4=lrelu
__global__ void gemm_kernel(const float* __restrict__ X, int ldx, int K,
                            const bf16* __restrict__ Wb, int ldw,
                            const float* __restrict__ bias,
                            const float* __restrict__ g,
                            const float* __restrict__ beta,
                            float* __restrict__ Y, int ldy,
                            int Mrows, int O, int flags) {
  int mt = blockIdx.x;
  int Kpad = ldw;
  extern __shared__ bf16 xs[];          // [16][Kpad] row tile in bf16
  int t = threadIdx.x;
  for (int i = t; i < 16 * Kpad; i += blockDim.x) {
    int r = i / Kpad, c = i % Kpad;
    int row = mt * 16 + r;
    float v = (row < Mrows && c < K) ? X[(long)row * ldx + c] : 0.f;
    xs[i] = (bf16)v;
  }
  __syncthreads();
  int wave = t >> 5, lane = t & 31;
  int ntiles = (O + 15) >> 4;
  int npairs = (ntiles + 1) >> 1;
  for (int p = blockIdx.y * 8 + wave; p < npairs; p += gridDim.y * 8) {
    int nt0 = 2 * p, nt1 = 2 * p + 1;
    v8f acc0 = {0.f,0.f,0.f,0.f,0.f,0.f,0.f,0.f};
    v8f acc1 = {0.f,0.f,0.f,0.f,0.f,0.f,0.f,0.f};
    int col0 = nt0 * 16 + (lane & 15);
    int col1 = nt1 * 16 + (lane & 15);
    const bf16* wcol0 = Wb + (long)((col0 < O) ? col0 : (O - 1)) * ldw;
    const bf16* wcol1 = Wb + (long)((col1 < O) ? col1 : (O - 1)) * ldw;
    const bf16* arow  = xs + (lane & 15) * Kpad;
    for (int kk = 0; kk < Kpad; kk += 32) {
      __builtin_prefetch(wcol0 + kk + 512, 0, 0);   // global_prefetch_b8
      v16bf a  = load_a_frag(arow + kk, lane);
      v16bf b0 = load_b_frag(wcol0 + kk, lane);
      v16bf b1 = load_b_frag(wcol1 + kk, lane);
      acc0 = wmma_bf16(a, b0, acc0);
      acc1 = wmma_bf16(a, b1, acc1);
    }
    int rbase = mt * 16 + ((lane < 16) ? 0 : 8);
#pragma unroll 2
    for (int h = 0; h < 2; h++) {
      int col = h ? col1 : col0;
      v8f* accp = h ? &acc1 : &acc0;
      if (col < O) {
        float add = (flags & 1) ? bias[col] : 0.f;
        float gs  = (flags & 2) ? g[col] * BN_SCALE_C : 1.f;
        float bb  = (flags & 2) ? beta[col] : 0.f;
#pragma unroll
        for (int i = 0; i < 8; i++) {
          int row = rbase + i;
          if (row < Mrows) {
            float v = ((*accp)[i] + add) * gs + bb;
            if (flags & 4) v = (v >= 0.f) ? v : 0.2f * v;
            Y[(long)row * ldy + col] = v;
          }
        }
      }
    }
  }
}

// ---- 7) global max + mean pool over N --------------------------------------
__global__ void pool_kernel(const float* __restrict__ E,
                            float* __restrict__ pooled, int N, int C) {
  int b = blockIdx.y;
  int c = blockIdx.x * blockDim.x + threadIdx.x;
  if (c >= C) return;
  const float* Eb = E + (long)b * N * C;
  float mx = NEG_INF_F, sm = 0.f;
  for (int n = 0; n < N; n++) {
    float v = Eb[(long)n * C + c];
    mx = fmaxf(mx, v);
    sm += v;
  }
  pooled[(long)b * 2 * C + c]     = mx;
  pooled[(long)b * 2 * C + C + c] = sm * (1.f / (float)N);
}

// ---- 8) concat env scalar ---------------------------------------------------
__global__ void concat_env_kernel(const float* __restrict__ m, int ldm,
                                  const float* __restrict__ env,
                                  float* __restrict__ mc, int EMB, int ld) {
  int b = blockIdx.x, t = threadIdx.x;
  for (int c = t; c < ld; c += blockDim.x) {
    float v = (c < EMB) ? m[(long)b * ldm + c] : ((c == EMB) ? env[b] : 0.f);
    mc[(long)b * ld + c] = v;
  }
}

// ---------------------------------------------------------------------------
static inline int rup32(int x) { return (x + 31) & ~31; }

extern "C" void kernel_launch(void* const* d_in, const int* in_sizes, int n_in,
                              void* d_out, int out_size, void* d_ws, size_t ws_size,
                              hipStream_t stream) {
  (void)in_sizes; (void)n_in; (void)out_size; (void)ws_size;
  const int B = 32, N = 1024, K = 20, EMB = 1024, OUT = 7500;

  const float* x   = (const float*)d_in[0];
  const float* env = (const float*)d_in[1];
  const float* Wl[5], *gl[5], *bl[5];
  for (int i = 0; i < 5; i++) {
    Wl[i] = (const float*)d_in[3 + 3 * i];
    gl[i] = (const float*)d_in[4 + 3 * i];
    bl[i] = (const float*)d_in[5 + 3 * i];
  }
  const float* W6  = (const float*)d_in[18];
  const float* g6  = (const float*)d_in[19];
  const float* b6  = (const float*)d_in[20];
  const float* Wm  = (const float*)d_in[21];
  const float* bm  = (const float*)d_in[22];
  const float* gm  = (const float*)d_in[23];
  const float* bem = (const float*)d_in[24];
  const float* Wd[5], *bd[5], *gd[5], *betad[5];
  for (int j = 0; j < 5; j++) {
    Wd[j]    = (const float*)d_in[25 + 4 * j + 0];
    bd[j]    = (const float*)d_in[25 + 4 * j + 1];
    gd[j]    = (const float*)d_in[25 + 4 * j + 2];
    betad[j] = (const float*)d_in[25 + 4 * j + 3];
  }
  const float* Wout = (const float*)d_in[45];
  const float* bout = (const float*)d_in[46];

  // ---- workspace partitioning ----
  char* ws = (char*)d_ws;
  size_t off = 0;
  auto alloc = [&](size_t bytes) -> void* {
    void* p = (void*)(ws + off);
    off += (bytes + 255) & ~(size_t)255;
    return p;
  };
  float* featcat = (float*)alloc((size_t)B * N * 1024 * 4); // concat x1..x5
  float* h0      = (float*)alloc((size_t)B * N * 21 * 4);
  bf16*  Hbf     = (bf16*) alloc((size_t)B * N * 512 * 2);
  float* gram    = (float*)alloc((size_t)B * N * N * 4);
  int*   knnidx  = (int*)  alloc((size_t)B * N * K * 4);
  float* ebuf    = (float*)alloc((size_t)B * N * 1024 * 4);
  float* pooled  = (float*)alloc((size_t)B * 2048 * 4);
  float* mbuf0   = (float*)alloc((size_t)B * 2048 * 4);
  float* mbuf2   = (float*)alloc((size_t)B * 2048 * 4);
  float* mcat    = (float*)alloc((size_t)B * 1056 * 4);

  const int Cin_l[5]  = {21, 64, 64, 128, 256};   // per-layer input width
  const int O_l[5]    = {64, 64, 128, 256, 512};
  const int slice[5]  = {0, 64, 128, 256, 512};
  bf16* Wbf_l[5];
  for (int i = 0; i < 5; i++)
    Wbf_l[i] = (bf16*)alloc((size_t)O_l[i] * rup32(2 * Cin_l[i]) * 2);
  bf16* W6bf  = (bf16*)alloc((size_t)EMB * 1024 * 2);
  bf16* Wmbf  = (bf16*)alloc((size_t)EMB * 2048 * 2);
  bf16* Wdbf[5];
  Wdbf[0] = (bf16*)alloc((size_t)2048 * 1056 * 2);
  for (int j = 1; j < 5; j++) Wdbf[j] = (bf16*)alloc((size_t)2048 * 2048 * 2);
  bf16* Woutbf = (bf16*)alloc((size_t)OUT * 2048 * 2);

  auto cvt = [&](const float* src, int ldsrc, bf16* dst, int lddst,
                 int rows, int cols) {
    long total = (long)rows * lddst;
    int blocks = (int)((total + 255) / 256);
    convert_pad_bf16<<<blocks, 256, 0, stream>>>(src, ldsrc, dst, lddst, rows, cols);
  };

  // ---- weight prep (f32 -> padded bf16, row stride = round32(K)) ----
  for (int i = 0; i < 5; i++)
    cvt(Wl[i], 2 * Cin_l[i], Wbf_l[i], rup32(2 * Cin_l[i]), O_l[i], 2 * Cin_l[i]);
  cvt(W6,   1024, W6bf,  1024, EMB, 1024);
  cvt(Wm,   2048, Wmbf,  2048, EMB, 2048);
  cvt(Wd[0], 1025, Wdbf[0], 1056, 2048, 1025);
  for (int j = 1; j < 5; j++) cvt(Wd[j], 2048, Wdbf[j], 2048, 2048, 2048);
  cvt(Wout, 2048, Woutbf, 2048, OUT, 2048);

  // ---- transpose input ----
  {
    long total = (long)B * 21 * N;
    transpose_x_kernel<<<(int)((total + 255) / 256), 256, 0, stream>>>(x, h0, B, 21, N);
  }

  // ---- 5 EdgeConv layers ----
  for (int l = 0; l < 5; l++) {
    const float* Hsrc = (l == 0) ? h0 : (featcat + slice[l - 1]);
    int ldh = (l == 0) ? 21 : 1024;
    int C    = Cin_l[l];
    int Cpad = rup32(C);
    int Kpw  = rup32(2 * C);
    // bf16 stage of H for Gram
    cvt(Hsrc, ldh, Hbf, Cpad, B * N, C);
    // Gram via WMMA: 4096 tiles/batch, 8 tiles per 256-thread block
    gram_kernel<<<dim3(512, B), 256, 0, stream>>>(Hbf, Cpad, gram, N, Cpad);
    // top-20 neighbors
    topk_kernel<<<B * N, 256, 0, stream>>>(gram, N, knnidx, K);
    // fused gather + GEMM + BN + lrelu + max-over-k
    int use_async = ((C & 3) == 0) ? 1 : 0;   // 16B-granule DMA path
    int rfbytes = (((K + 1) * C * 4) + 31) & ~31;
    size_t smem = (size_t)rfbytes + (size_t)32 * Kpw * 2;
    edgeconv_kernel<<<B * N, 256, smem, stream>>>(
        Hsrc, ldh, C, knnidx, K, Wbf_l[l], Kpw, O_l[l], gl[l], bl[l],
        featcat + slice[l], 1024, N, rfbytes, use_async);
  }

  // ---- embedding GEMM: e = lrelu(bn(cat @ W6^T)) ----  (64 tiles -> 32 pairs)
  gemm_kernel<<<dim3((B * N) / 16, 4), 256, (size_t)16 * 1024 * 2, stream>>>(
      featcat, 1024, 1024, W6bf, 1024, nullptr, g6, b6,
      ebuf, 1024, B * N, EMB, /*bn|lrelu*/ 6);

  // ---- pooling: [max, mean] over N -> (B, 2048) ----
  pool_kernel<<<dim3(4, B), 256, 0, stream>>>(ebuf, pooled, N, 1024);

  // ---- m = lrelu(bn(pooled @ Wm^T + bm)) ----
  gemm_kernel<<<dim3(2, 4), 256, (size_t)16 * 2048 * 2, stream>>>(
      pooled, 2048, 2048, Wmbf, 2048, bm, gm, bem,
      mbuf0, 1024, B, EMB, /*bias|bn|lrelu*/ 7);

  // ---- concat env -> (B, 1025) padded to 1056 ----
  concat_env_kernel<<<B, 256, 0, stream>>>(mbuf0, 1024, env, mcat, EMB, 1056);

  // ---- decoder: 5 x lrelu(bn(m @ Wd^T + bd)) ----  (128 tiles -> 64 pairs)
  const float* dx = mcat; int dldx = 1056, dK = 1025;
  float* dy = mbuf0;
  for (int j = 0; j < 5; j++) {
    int Kpw = (j == 0) ? 1056 : 2048;
    gemm_kernel<<<dim3(2, 8), 256, (size_t)16 * Kpw * 2, stream>>>(
        dx, dldx, dK, Wdbf[j], Kpw, bd[j], gd[j], betad[j],
        dy, 2048, B, 2048, /*bias|bn|lrelu*/ 7);
    dx = dy; dldx = 2048; dK = 2048;
    dy = (dy == mbuf0) ? mbuf2 : mbuf0;
  }

  // ---- output: out = m @ Wout^T + bout ----  (469 tiles -> 235 pairs)
  gemm_kernel<<<dim3(2, 8), 256, (size_t)16 * 2048 * 2, stream>>>(
      dx, 2048, 2048, Woutbf, 2048, bout, nullptr, nullptr,
      (float*)d_out, OUT, B, OUT, /*bias*/ 1);
}
